// MultiheadedAttention_40467181863499
// MI455X (gfx1250) — matrix operand order repaired and checked by reading, not compiled, not run on hardware
//
#include <hip/hip_runtime.h>
#include <hip/hip_bf16.h>

// Problem constants (from reference)
#define BB 2
#define SS 2048
#define DD 1024
#define HH 16
#define HD 64
#define D3 3072   // 3*D
#define MROWS (BB*SS)   // 4096

typedef unsigned short u16;
typedef __attribute__((ext_vector_type(16))) __bf16 v16bf;
typedef __attribute__((ext_vector_type(8)))  float  v8f;
typedef __attribute__((ext_vector_type(4)))  unsigned int u32x4;

union ABFrag { v16bf v; u32x4 q[2]; };

__device__ __forceinline__ u16 f2bf(float f) {
  unsigned u = __builtin_bit_cast(unsigned, f);
  u += 0x7FFFu + ((u >> 16) & 1u);          // round-to-nearest-even
  return (u16)(u >> 16);
}

__device__ __forceinline__ v8f vzero8() {
  v8f z;
#pragma unroll
  for (int i = 0; i < 8; ++i) z[i] = 0.0f;
  return z;
}

__device__ __forceinline__ v8f wmma_bf16(v16bf a, v16bf b, v8f c) {
  // D = A(16x32 bf16) * B(32x16 bf16) + C(16x16 f32)
  return __builtin_amdgcn_wmma_f32_16x16x32_bf16(
      /*neg_a=*/false, a, /*neg_b=*/false, b,
      /*c_mod=*/(short)0, c, /*reuse_a=*/false, /*reuse_b=*/false);
}

// A fragment: rows 0..15 of a row-major [*, ld] bf16 matrix, contraction offset kk.
// A layout: lane L holds row M=L&15, contraction elements K = (L>>4)*16 + 0..15
// (contiguous, matching load_b16t).
__device__ __forceinline__ v16bf load_a16(const u16* base, int ld, int lane, int kk) {
  const int m  = lane & 15;
  const int kh = (lane >> 4) << 4;
  ABFrag f;
  const u16* p = base + (size_t)m * ld + kk + kh;
  f.q[0] = *(const u32x4*)(p);
  f.q[1] = *(const u32x4*)(p + 8);
  return f.v;
}

// B fragment from an N-major ("B transposed") matrix BT[n][k]: lane L holds
// column N=L&15, contraction elements K = (L>>4)*16 + 0..15 (contiguous).
__device__ __forceinline__ v16bf load_b16t(const u16* base, int ld, int lane, int kk) {
  const int n  = lane & 15;
  const int kb = (lane >> 4) << 4;
  ABFrag f;
  const u16* p = base + (size_t)n * ld + kk + kb;
  f.q[0] = *(const u32x4*)(p);
  f.q[1] = *(const u32x4*)(p + 8);
  return f.v;
}

// ---------------------------------------------------------------- converts
__global__ void __launch_bounds__(256)
cvt_bf16_kernel(const float* __restrict__ src, u16* __restrict__ dst, int n) {
  int i = blockIdx.x * blockDim.x + threadIdx.x;
  const int stride = gridDim.x * blockDim.x;
  for (; i < n; i += stride) dst[i] = f2bf(src[i]);
}

// src is [K][N] row-major f32; dst is [N][K] bf16 (N-major, contraction contiguous)
__global__ void __launch_bounds__(256)
cvt_bf16_T_kernel(const float* __restrict__ src, u16* __restrict__ dst, int N, int K) {
  int i = blockIdx.x * blockDim.x + threadIdx.x;
  const int stride = gridDim.x * blockDim.x;
  const int total = N * K;
  for (; i < total; i += stride) {
    const int n = i / K;
    const int k = i - n * K;
    dst[i] = f2bf(src[(size_t)k * N + n]);
  }
}

// ---------------------------------------------------- QKV GEMM + bias + scatter
// C[M=4096, N=3072] = xb[M,K=1024] @ Wqkv[K,N] + bqkv; scatter to
//   Q[b,h,s,d], K[b,h,s,d] (row-major over d) and Vt[b,h,d,s] (keys contiguous).
__global__ void __launch_bounds__(256)
qkv_gemm_kernel(const u16* __restrict__ xb, const u16* __restrict__ WqkvT,
                const float* __restrict__ bqkv,
                u16* __restrict__ Qb, u16* __restrict__ Kb, u16* __restrict__ Vt) {
  const int lane = threadIdx.x & 31;
  const int wave = threadIdx.x >> 5;
  const int lh   = lane & 15;
  const int kh8  = (lane >> 4) << 3;

  const int tid = blockIdx.x * 8 + wave;         // 12288 wave-tiles (128 x 96)
  const int tn  = tid % (D3 / 32);
  const int tm  = tid / (D3 / 32);
  const int m0  = tm * 32;
  const int n0  = tn * 32;

  v8f acc[2][2];
#pragma unroll
  for (int i = 0; i < 2; ++i)
#pragma unroll
    for (int j = 0; j < 2; ++j) acc[i][j] = vzero8();

  const u16* A0 = xb + (size_t)m0 * DD;
  const u16* A1 = xb + (size_t)(m0 + 16) * DD;
  const u16* B0 = WqkvT + (size_t)n0 * DD;
  const u16* B1 = WqkvT + (size_t)(n0 + 16) * DD;

  for (int kk = 0; kk < DD; kk += 32) {
    const v16bf a0 = load_a16(A0, DD, lane, kk);
    const v16bf a1 = load_a16(A1, DD, lane, kk);
    const v16bf b0 = load_b16t(B0, DD, lane, kk);
    const v16bf b1 = load_b16t(B1, DD, lane, kk);
    acc[0][0] = wmma_bf16(a0, b0, acc[0][0]);
    acc[0][1] = wmma_bf16(a0, b1, acc[0][1]);
    acc[1][0] = wmma_bf16(a1, b0, acc[1][0]);
    acc[1][1] = wmma_bf16(a1, b1, acc[1][1]);
  }

#pragma unroll
  for (int j = 0; j < 2; ++j) {
    const int col = n0 + j * 16 + lh;            // 0..3071
    const float bias = bqkv[col];
    const int h    = col / 192;                  // head (interleaved q/k/v split)
    const int w    = col - h * 192;
    const int kind = w >> 6;                     // 0=q 1=k 2=v (uniform across the 16-span)
    const int d    = w & 63;
#pragma unroll
    for (int i = 0; i < 2; ++i) {
#pragma unroll
      for (int r = 0; r < 8; ++r) {
        const int row = m0 + i * 16 + r + kh8;   // 0..4095
        const float v = acc[i][j][r] + bias;
        const int b = row >> 11;                 // /2048
        const int s = row & (SS - 1);
        const size_t bh = (size_t)(b * HH + h);
        const u16 bv = f2bf(v);
        if (kind == 0)      Qb[(bh * SS + s) * HD + d] = bv;
        else if (kind == 1) Kb[(bh * SS + s) * HD + d] = bv;
        else                Vt[(bh * HD + d) * SS + s] = bv;
      }
    }
  }
}

// ------------------------------------------------------------- flash attention
// One wave owns 16 queries of one (b,h); key tiles of 32 with online softmax.
__global__ void __launch_bounds__(256)
attn_kernel(const u16* __restrict__ Qb, const u16* __restrict__ Kb,
            const u16* __restrict__ Vt, u16* __restrict__ vals) {
  __shared__ __align__(16) u16 pbuf[8][16 * 32];   // per-wave P staging (C->A layout)
  const int lane = threadIdx.x & 31;
  const int wave = threadIdx.x >> 5;
  const int lh   = lane & 15;
  const int kh8  = (lane >> 4) << 3;

  const int bh = blockIdx.x >> 4;                  // 32 (b,h) pairs
  const int qt = blockIdx.x & 15;                  // 16 query tiles of 128
  const int q0 = qt * 128 + wave * 16;
  const int h  = bh & (HH - 1);
  const int b  = bh >> 4;

  const u16* Qp = Qb + ((size_t)bh * SS + q0) * HD;
  const u16* Kp = Kb + (size_t)bh * SS * HD;
  const u16* Vp = Vt + (size_t)bh * HD * SS;
  u16* P = pbuf[wave];

  // Q fragments held in registers for the whole key loop (16x64 -> 2 A frags)
  const v16bf qa0 = load_a16(Qp, HD, lane, 0);
  const v16bf qa1 = load_a16(Qp, HD, lane, 32);

  v8f o[4];
#pragma unroll
  for (int t = 0; t < 4; ++t) o[t] = vzero8();
  float mrun[8], lrun[8];
#pragma unroll
  for (int r = 0; r < 8; ++r) { mrun[r] = -1e30f; lrun[r] = 0.0f; }

  for (int kt = 0; kt < SS; kt += 32) {
    if (kt + 32 < SS)
      __builtin_prefetch(Kp + (size_t)(kt + 32) * HD + lane * 16, 0, 0);

    // scores: two 16x16 tiles (keys kt..kt+15 and kt+16..kt+31), K-dim = 64
    v8f s0 = vzero8(), s1 = vzero8();
    s0 = wmma_bf16(qa0, load_b16t(Kp + (size_t)kt * HD, HD, lane, 0),  s0);
    s0 = wmma_bf16(qa1, load_b16t(Kp + (size_t)kt * HD, HD, lane, 32), s0);
    s1 = wmma_bf16(qa0, load_b16t(Kp + (size_t)(kt + 16) * HD, HD, lane, 0),  s1);
    s1 = wmma_bf16(qa1, load_b16t(Kp + (size_t)(kt + 16) * HD, HD, lane, 32), s1);

    float tmax[8];
#pragma unroll
    for (int r = 0; r < 8; ++r) {
      s0[r] *= 0.125f;                           // 1/sqrt(64)
      s1[r] *= 0.125f;
      tmax[r] = fmaxf(s0[r], s1[r]);
    }
    // row-wise max across the 16 lanes of each half (C layout keeps a row per half)
#pragma unroll
    for (int dlt = 1; dlt < 16; dlt <<= 1)
#pragma unroll
      for (int r = 0; r < 8; ++r) tmax[r] = fmaxf(tmax[r], __shfl_xor(tmax[r], dlt, 32));

    float fac[8], rs[8];
#pragma unroll
    for (int r = 0; r < 8; ++r) {
      const float mn = fmaxf(mrun[r], tmax[r]);
      fac[r] = __expf(mrun[r] - mn);
      mrun[r] = mn;
      const float p0 = __expf(s0[r] - mn);
      const float p1 = __expf(s1[r] - mn);
      rs[r] = p0 + p1;
      const int row = r + kh8;
      P[row * 32 + lh]      = f2bf(p0);          // C layout -> row-major 16x32 in LDS
      P[row * 32 + 16 + lh] = f2bf(p1);
    }
#pragma unroll
    for (int dlt = 1; dlt < 16; dlt <<= 1)
#pragma unroll
      for (int r = 0; r < 8; ++r) rs[r] += __shfl_xor(rs[r], dlt, 32);
#pragma unroll
    for (int r = 0; r < 8; ++r) lrun[r] = lrun[r] * fac[r] + rs[r];

#pragma unroll
    for (int t = 0; t < 4; ++t)
#pragma unroll
      for (int r = 0; r < 8; ++r) o[t][r] *= fac[r];

    // reload P as an A fragment (ds_load_b128 x2), then P @ V over 4 dim tiles
    const v16bf pa = load_a16(P, 32, lane, 0);
#pragma unroll
    for (int t = 0; t < 4; ++t)
      o[t] = wmma_bf16(pa, load_b16t(Vp + (size_t)t * 16 * SS, SS, lane, kt), o[t]);
  }

  // normalize and write vals[b, s, h*64 + d] (bf16, row-major [B*S, D])
#pragma unroll
  for (int t = 0; t < 4; ++t) {
#pragma unroll
    for (int r = 0; r < 8; ++r) {
      const int row = q0 + r + kh8;
      const int col = h * HD + t * 16 + lh;
      const float v = o[t][r] / lrun[r];
      vals[((size_t)b * SS + row) * DD + col] = f2bf(v);
    }
  }
}

// ----------------------------------------------------- output projection GEMM
__global__ void __launch_bounds__(256)
out_gemm_kernel(const u16* __restrict__ valsb, const u16* __restrict__ WoT,
                const float* __restrict__ bo, float* __restrict__ out) {
  const int lane = threadIdx.x & 31;
  const int wave = threadIdx.x >> 5;
  const int lh   = lane & 15;
  const int kh8  = (lane >> 4) << 3;

  const int tid = blockIdx.x * 8 + wave;          // 4096 wave-tiles (128 x 32)
  const int tn  = tid % (DD / 32);
  const int tm  = tid / (DD / 32);
  const int m0  = tm * 32;
  const int n0  = tn * 32;

  v8f acc[2][2];
#pragma unroll
  for (int i = 0; i < 2; ++i)
#pragma unroll
    for (int j = 0; j < 2; ++j) acc[i][j] = vzero8();

  const u16* A0 = valsb + (size_t)m0 * DD;
  const u16* A1 = valsb + (size_t)(m0 + 16) * DD;
  const u16* B0 = WoT + (size_t)n0 * DD;
  const u16* B1 = WoT + (size_t)(n0 + 16) * DD;

  for (int kk = 0; kk < DD; kk += 32) {
    const v16bf a0 = load_a16(A0, DD, lane, kk);
    const v16bf a1 = load_a16(A1, DD, lane, kk);
    const v16bf b0 = load_b16t(B0, DD, lane, kk);
    const v16bf b1 = load_b16t(B1, DD, lane, kk);
    acc[0][0] = wmma_bf16(a0, b0, acc[0][0]);
    acc[0][1] = wmma_bf16(a0, b1, acc[0][1]);
    acc[1][0] = wmma_bf16(a1, b0, acc[1][0]);
    acc[1][1] = wmma_bf16(a1, b1, acc[1][1]);
  }

#pragma unroll
  for (int j = 0; j < 2; ++j) {
    const int col = n0 + j * 16 + lh;
    const float bias = bo[col];
#pragma unroll
    for (int i = 0; i < 2; ++i)
#pragma unroll
      for (int r = 0; r < 8; ++r) {
        const int row = m0 + i * 16 + r + kh8;
        out[(size_t)row * DD + col] = acc[i][j][r] + bias;
      }
  }
}

// --------------------------------------------------------------------- launch
extern "C" void kernel_launch(void* const* d_in, const int* in_sizes, int n_in,
                              void* d_out, int out_size, void* d_ws, size_t ws_size,
                              hipStream_t stream) {
  (void)in_sizes; (void)n_in; (void)out_size; (void)ws_size;
  const float* x    = (const float*)d_in[0];   // [B,S,D]
  const float* Wqkv = (const float*)d_in[1];   // [D,3D]
  const float* bqkv = (const float*)d_in[2];   // [3D]
  const float* Wo   = (const float*)d_in[3];   // [D,D]
  const float* bo   = (const float*)d_in[4];   // [D]
  float* out = (float*)d_out;                  // [B,S,D] f32

  char* ws = (char*)d_ws;                      // ~48 MB of bf16 scratch
  u16* xb    = (u16*)(ws + (size_t)0);              // 8 MB  [4096,1024]
  u16* WqkvT = (u16*)(ws + ((size_t)8  << 20));     // 6 MB  [3072,1024] (N-major)
  u16* WoT   = (u16*)(ws + ((size_t)14 << 20));     // 2 MB  [1024,1024] (N-major)
  u16* Qb    = (u16*)(ws + ((size_t)16 << 20));     // 8 MB  [B,H,S,64]
  u16* Kb    = (u16*)(ws + ((size_t)24 << 20));     // 8 MB  [B,H,S,64]
  u16* Vt    = (u16*)(ws + ((size_t)32 << 20));     // 8 MB  [B,H,64,S]
  u16* vals  = (u16*)(ws + ((size_t)40 << 20));     // 8 MB  [4096,1024]

  cvt_bf16_kernel<<<2048, 256, 0, stream>>>(x, xb, MROWS * DD);
  cvt_bf16_T_kernel<<<2048, 256, 0, stream>>>(Wqkv, WqkvT, D3, DD);
  cvt_bf16_T_kernel<<<1024, 256, 0, stream>>>(Wo, WoT, DD, DD);
  qkv_gemm_kernel<<<(MROWS / 32) * (D3 / 32) / 8, 256, 0, stream>>>(xb, WqkvT, bqkv, Qb, Kb, Vt);
  attn_kernel<<<BB * HH * (SS / 128), 256, 0, stream>>>(Qb, Kb, Vt, vals);
  out_gemm_kernel<<<(MROWS / 32) * (DD / 32) / 8, 256, 0, stream>>>(vals, WoT, bo, out);
}
